// Layer54_39290360824207
// MI455X (gfx1250) — matrix operand order repaired and checked by reading, not compiled, not run on hardware
//
#include <hip/hip_runtime.h>

#define BB 2
#define SS 2048
#define HH 1024
#define NH 16
#define HD 64

typedef __attribute__((ext_vector_type(16))) __bf16 v16bf;
typedef __attribute__((ext_vector_type(8)))  __bf16 v8bf;
typedef __attribute__((ext_vector_type(8)))  float  v8f;
typedef __attribute__((ext_vector_type(4)))  float  v4f;

#define WMMA(acc, a, b)                                                  \
  acc = __builtin_amdgcn_wmma_f32_16x16x32_bf16(false, a, false, b,      \
                                                (short)0, acc, false, false)

static __device__ inline v16bf join8(v8bf lo, v8bf hi) {
  v16bf r;
#pragma unroll
  for (int e = 0; e < 8; ++e) { r[e] = lo[e]; r[8 + e] = hi[e]; }
  return r;
}

// A-fragment (16x32 bf16): lane = row; 16 contiguous K (k = 16*half + e).
static __device__ inline v16bf loadA(const __bf16* row, int k0, int aoff) {
  return join8(*(const v8bf*)(row + k0 + 2 * aoff),
               *(const v8bf*)(row + k0 + 2 * aoff + 8));
}
#define LOADB(p) (*(const v16bf*)(p))

// -------------------------------------------------------------------------
// Kernel 0: fp32 -> bf16 conversion pre-pass (bandwidth bound).
// -------------------------------------------------------------------------
__global__ __launch_bounds__(256)
void cvt_bf16_kernel(const float* __restrict__ src, __bf16* __restrict__ dst,
                     int n8)
{
  int i = blockIdx.x * 256 + threadIdx.x;
  if (i >= n8) return;
  const v4f* s = (const v4f*)(src + (size_t)i * 8);
  v4f x0 = s[0], x1 = s[1];
  v8bf o;
#pragma unroll
  for (int e = 0; e < 4; ++e) { o[e] = (__bf16)x0[e]; o[4 + e] = (__bf16)x1[e]; }
  *(v8bf*)(dst + (size_t)i * 8) = o;
}

// -------------------------------------------------------------------------
// Kernel 1: QKV GEMM, all-bf16 operands, register double-buffered.
// One wave -> 32x64 output tile of one of {Q,K,V}; 8 WMMAs per k-step.
// Q,K -> [B,NH,S,HD]; V -> transposed [B,NH,HD,S] (epilogue stores packed).
// -------------------------------------------------------------------------
__global__ __launch_bounds__(256)
void qkv_gemm_kernel(const __bf16* __restrict__ Xb, const __bf16* __restrict__ Wb,
                     const float* __restrict__ bq, const float* __restrict__ bk,
                     const float* __restrict__ bv,
                     __bf16* __restrict__ Qh, __bf16* __restrict__ Kh,
                     __bf16* __restrict__ Vt)
{
  const int lane = threadIdx.x & 31;
  const int wave = threadIdx.x >> 5;
  const int lr   = lane & 15;
  const int hi   = lane >> 4;
  const int aoff = hi ? 8 : 0;
  const int boff = hi ? 16 : 0;

  const int MT = (BB * SS) / 32;   // 128
  const int NT = HH / 64;          // 16

  int wg    = blockIdx.x * 8 + wave;           // 0 .. 3*128*16-1
  int which = wg / (MT * NT);
  int rem   = wg % (MT * NT);
  int mbase = (rem / NT) * 32;
  int nbase = (rem % NT) * 64;

  const __bf16* W    = Wb + (size_t)which * HH * HH;
  const float*  bias = (which == 0) ? bq : (which == 1) ? bk : bv;

  const __bf16* Xr0 = Xb + (size_t)(mbase + lr) * HH;
  const __bf16* Xr1 = Xb + (size_t)(mbase + 16 + lr) * HH;
  const __bf16* Wr  = W + (size_t)(nbase + lr) * HH + boff;

  v8f acc[8] = {};

  // double-buffered fragments
  v16bf a0c, a1c, b0c, b1c, b2c, b3c;
  v16bf a0n, a1n, b0n, b1n, b2n, b3n;

  a0c = loadA(Xr0, 0, aoff);
  a1c = loadA(Xr1, 0, aoff);
  b0c = LOADB(Wr);
  b1c = LOADB(Wr + 16 * HH);
  b2c = LOADB(Wr + 32 * HH);
  b3c = LOADB(Wr + 48 * HH);

  for (int ks = 0; ks < 32; ks += 2) {
    const int kn = (ks + 1) * 32;
    a0n = loadA(Xr0, kn, aoff);
    a1n = loadA(Xr1, kn, aoff);
    b0n = LOADB(Wr + kn);
    b1n = LOADB(Wr + 16 * HH + kn);
    b2n = LOADB(Wr + 32 * HH + kn);
    b3n = LOADB(Wr + 48 * HH + kn);

    WMMA(acc[0], a0c, b0c); WMMA(acc[1], a0c, b1c);
    WMMA(acc[2], a0c, b2c); WMMA(acc[3], a0c, b3c);
    WMMA(acc[4], a1c, b0c); WMMA(acc[5], a1c, b1c);
    WMMA(acc[6], a1c, b2c); WMMA(acc[7], a1c, b3c);

    if (ks + 2 < 32) {
      const int k2 = (ks + 2) * 32;
      a0c = loadA(Xr0, k2, aoff);
      a1c = loadA(Xr1, k2, aoff);
      b0c = LOADB(Wr + k2);
      b1c = LOADB(Wr + 16 * HH + k2);
      b2c = LOADB(Wr + 32 * HH + k2);
      b3c = LOADB(Wr + 48 * HH + k2);
    }

    WMMA(acc[0], a0n, b0n); WMMA(acc[1], a0n, b1n);
    WMMA(acc[2], a0n, b2n); WMMA(acc[3], a0n, b3n);
    WMMA(acc[4], a1n, b0n); WMMA(acc[5], a1n, b1n);
    WMMA(acc[6], a1n, b2n); WMMA(acc[7], a1n, b3n);
  }

  // ---- epilogue: bias, cast, scatter into attention layouts ----
  if (which < 2) {
    __bf16* dst = (which == 0) ? Qh : Kh;
#pragma unroll
    for (int j = 0; j < 4; ++j) {
      const int n = nbase + j * 16 + lr;
      const float bval = bias[n];
      const int hcol = n >> 6;
      const int d    = n & 63;
#pragma unroll
      for (int g = 0; g < 2; ++g) {
#pragma unroll
        for (int r = 0; r < 8; ++r) {
          const int m  = mbase + g * 16 + r + (hi ? 8 : 0);
          const int b_ = m >> 11;
          const int s  = m & (SS - 1);
          dst[((size_t)(b_ * NH + hcol) * SS + s) * HD + d] =
              (__bf16)(acc[g * 4 + j][r] + bval);
        }
      }
    }
  } else {
    // V transposed: for fixed (lane, j, g) the 8 accumulator rows hit 8
    // CONTIGUOUS s addresses -> pack into one 16-byte store.
#pragma unroll
    for (int j = 0; j < 4; ++j) {
      const int n = nbase + j * 16 + lr;
      const float bval = bias[n];
      const int hcol = n >> 6;
      const int d    = n & 63;
#pragma unroll
      for (int g = 0; g < 2; ++g) {
        const int m0 = mbase + g * 16 + (hi ? 8 : 0);   // first of 8 rows
        const int b_ = m0 >> 11;
        const int s0 = m0 & (SS - 1);
        v8bf pk;
#pragma unroll
        for (int r = 0; r < 8; ++r) pk[r] = (__bf16)(acc[g * 4 + j][r] + bval);
        *(v8bf*)(Vt + ((size_t)(b_ * NH + hcol) * HD + d) * SS + s0) = pk;
      }
    }
  }
}

// -------------------------------------------------------------------------
// Kernel 2: flash attention.  One wave -> 16-row Q tile of one (b,h).
// All loads hoisted to top of each 32-key block; row-sum via P @ ones WMMA.
// -------------------------------------------------------------------------
__global__ __launch_bounds__(256)
void attn_kernel(const __bf16* __restrict__ Qh, const __bf16* __restrict__ Kh,
                 const __bf16* __restrict__ Vt, const float* __restrict__ mask,
                 float* __restrict__ Out)
{
  __shared__ __bf16 ldsP[8][16][32];

  const int lane = threadIdx.x & 31;
  const int wave = threadIdx.x >> 5;
  const int lr   = lane & 15;
  const int hi   = lane >> 4;
  const int aoff = hi ? 8 : 0;
  const int boff = hi ? 16 : 0;

  const int QT = SS / 16;            // 128
  int wg  = blockIdx.x * 8 + wave;
  int b   = wg / (NH * QT);
  int rem = wg % (NH * QT);
  int h   = rem / QT;
  int qbase = (rem % QT) * 16;

  const __bf16* Qbase = Qh + (size_t)(b * NH + h) * SS * HD;
  const __bf16* Kbase = Kh + (size_t)(b * NH + h) * SS * HD;
  const __bf16* Vbase = Vt + (size_t)(b * NH + h) * HD * SS;
  const float*  mline = mask + (size_t)b * SS;

  const __bf16* qr = Qbase + (size_t)(qbase + lr) * HD;
  v16bf qa0 = loadA(qr, 0, aoff);
  v16bf qa1 = loadA(qr, 32, aoff);

  v16bf ones;
#pragma unroll
  for (int e = 0; e < 16; ++e) ones[e] = (__bf16)1.0f;

  v8f o[4] = {};
  v8f lacc = {};                     // row sums, maintained by P @ ones WMMA
  float mrow[8];
#pragma unroll
  for (int r = 0; r < 8; ++r) mrow[r] = -1e30f;

  const float sc = 0.125f;           // 1/sqrt(64)

  for (int kb = 0; kb < SS / 32; ++kb) {
    const int k0 = kb * 32;

    // ---- hoisted loads for this block (clause + latency hiding) ----
    const __bf16* kr0 = Kbase + (size_t)(k0 + lr) * HD + boff;
    const __bf16* kr1 = Kbase + (size_t)(k0 + 16 + lr) * HD + boff;
    v16bf kf00 = LOADB(kr0);
    v16bf kf01 = LOADB(kr0 + 32);
    v16bf kf10 = LOADB(kr1);
    v16bf kf11 = LOADB(kr1 + 32);
    v16bf vf0 = LOADB(Vbase + (size_t)(lr) * SS + k0 + boff);
    v16bf vf1 = LOADB(Vbase + (size_t)(16 + lr) * SS + k0 + boff);
    v16bf vf2 = LOADB(Vbase + (size_t)(32 + lr) * SS + k0 + boff);
    v16bf vf3 = LOADB(Vbase + (size_t)(48 + lr) * SS + k0 + boff);
    const float mk0 = mline[k0 + lr];
    const float mk1 = mline[k0 + 16 + lr];

    if (kb + 1 < SS / 32) {
      __builtin_prefetch(Kbase + (size_t)(k0 + 32 + lr) * HD, 0, 1);
      __builtin_prefetch(Vbase + (size_t)lr * SS + k0 + 32, 0, 1);
    }

    // ---- scores: two 16x16 tiles ----
    v8f s0 = {}, s1 = {};
    WMMA(s0, qa0, kf00); WMMA(s0, qa1, kf01);
    WMMA(s1, qa0, kf10); WMMA(s1, qa1, kf11);

    // ---- online softmax: max via lane-half butterflies ----
    float p0[8], p1[8];
#pragma unroll
    for (int r = 0; r < 8; ++r) {
      float a0 = s0[r] * sc + mk0;
      float a1 = s1[r] * sc + mk1;
      float t = fmaxf(a0, a1);
      t = fmaxf(t, __shfl_xor(t, 1, 32));
      t = fmaxf(t, __shfl_xor(t, 2, 32));
      t = fmaxf(t, __shfl_xor(t, 4, 32));
      t = fmaxf(t, __shfl_xor(t, 8, 32));
      const float mn   = fmaxf(mrow[r], t);
      const float corr = __expf(mrow[r] - mn);
      mrow[r] = mn;
      p0[r] = __expf(a0 - mn);
      p1[r] = __expf(a1 - mn);
      lacc[r] *= corr;
      o[0][r] *= corr; o[1][r] *= corr; o[2][r] *= corr; o[3][r] *= corr;
    }

    // ---- P: C-layout -> bf16 A-layout via per-wave LDS tile ----
    __bf16* lp = &ldsP[wave][0][0];
#pragma unroll
    for (int r = 0; r < 8; ++r) {
      const int row = r + (hi ? 8 : 0);
      lp[row * 32 + lr]      = (__bf16)p0[r];
      lp[row * 32 + 16 + lr] = (__bf16)p1[r];
    }
    v16bf pf = loadA(lp + lr * 32, 0, aoff);

    // ---- row-sum via WMMA (P @ ones), context accumulate ----
    WMMA(lacc, pf, ones);
    WMMA(o[0], pf, vf0);
    WMMA(o[1], pf, vf1);
    WMMA(o[2], pf, vf2);
    WMMA(o[3], pf, vf3);
  }

  // ---- normalize and write [B,S,H] fp32 ----
#pragma unroll
  for (int r = 0; r < 8; ++r) {
    const float inv = 1.0f / lacc[r];
    const int s = qbase + r + (hi ? 8 : 0);
    float* dst = Out + ((size_t)b * SS + s) * HH + h * HD;
    dst[lr]      = o[0][r] * inv;
    dst[16 + lr] = o[1][r] * inv;
    dst[32 + lr] = o[2][r] * inv;
    dst[48 + lr] = o[3][r] * inv;
  }
}

// -------------------------------------------------------------------------
extern "C" void kernel_launch(void* const* d_in, const int* in_sizes, int n_in,
                              void* d_out, int out_size, void* d_ws, size_t ws_size,
                              hipStream_t stream)
{
  const float* X    = (const float*)d_in[0];
  const float* mask = (const float*)d_in[1];
  const float* Wq   = (const float*)d_in[2];
  const float* bq   = (const float*)d_in[3];
  const float* Wk   = (const float*)d_in[4];
  const float* bk   = (const float*)d_in[5];
  const float* Wv   = (const float*)d_in[6];
  const float* bv   = (const float*)d_in[7];

  char* ws = (char*)d_ws;
  const size_t x_elems  = (size_t)BB * SS * HH;         // 4M
  const size_t w_elems  = (size_t)HH * HH;              // 1M
  const size_t qkv_elems = (size_t)BB * NH * SS * HD;   // 4M

  __bf16* Xb = (__bf16*)(ws);
  __bf16* Wb = (__bf16*)(ws + x_elems * 2);                    // [3][H][H]
  __bf16* Qh = (__bf16*)(ws + (x_elems + 3 * w_elems) * 2);
  __bf16* Kh = Qh + qkv_elems;
  __bf16* Vt = Kh + qkv_elems;

  // --- pre-convert activations and weights to bf16 ---
  {
    int n8 = (int)(x_elems / 8);
    cvt_bf16_kernel<<<(n8 + 255) / 256, 256, 0, stream>>>(X, Xb, n8);
    n8 = (int)(w_elems / 8);
    cvt_bf16_kernel<<<(n8 + 255) / 256, 256, 0, stream>>>(Wq, Wb, n8);
    cvt_bf16_kernel<<<(n8 + 255) / 256, 256, 0, stream>>>(Wk, Wb + w_elems, n8);
    cvt_bf16_kernel<<<(n8 + 255) / 256, 256, 0, stream>>>(Wv, Wb + 2 * w_elems, n8);
  }

  // --- QKV GEMM: 3 * 128 m-tiles * 16 n-tiles = 6144 waves / 8 per block ---
  qkv_gemm_kernel<<<768, 256, 0, stream>>>(Xb, Wb, bq, bk, bv, Qh, Kh, Vt);

  // --- attention: 2*16*128 = 4096 q-tile waves / 8 per block ---
  attn_kernel<<<512, 256, 0, stream>>>(Qh, Kh, Vt, mask, (float*)d_out);
}
